// CPF_prop_f_87144886436370
// MI455X (gfx1250) — matrix operand order repaired and checked by reading, not compiled, not run on hardware
//
#include <hip/hip_runtime.h>
#include <hip/hip_bf16.h>
#include <math.h>

// N=262144 rows, C=128 features, P=8 clusters.
// Roofline: 8.6 GFLOP vs 256MB mandatory HBM traffic -> ~11us floor at 23.3TB/s;
// needs ~780 TFLOP/s effective GEMM rate, so use bf16 WMMA with split-bf16
// (xh*Wh + xl*Wh + xh*Wl) for ~f32 accuracy at matrix-core speed.
// Rows are binned by cluster so each 128-row block does ONE dense GEMM (no 8x
// masked-GEMM overhead). W planes are pre-split/packed in WMMA B-operand order
// and staged to LDS via async global->LDS loads (ASYNCcnt) overlapped with the
// per-row normalization pass.

#define CDIM  128
#define PCLUS 8

typedef __attribute__((ext_vector_type(16))) __bf16 v16bf;
typedef __attribute__((ext_vector_type(8)))  __bf16 v8bf;
typedef __attribute__((ext_vector_type(8)))  float  v8f;
typedef __attribute__((ext_vector_type(4)))  int    v4i;
typedef __attribute__((address_space(1))) v4i* g_v4i_p;   // global (HIP __device__) int4*
typedef __attribute__((address_space(3))) v4i* l_v4i_p;   // LDS (__shared__) int4*

// ---- workspace layout (bytes) ----
// [0, 524288): per cluster c: Wh plane (32KB) at c*65536, Wl plane at c*65536+32768
//   plane chunk layout: byte ((ks*2 + h)*128 + n)*32 holds 16 bf16 = W[K][n],
//   K = ks*32 + 16*h + e  (e=0..15)  -> the per-lane B operand of
//   v_wmma_f32_16x16x32_bf16 in contiguous K order, matching A (lanes 0-15: K 0-15, lanes 16-31: K 16-31).
// ctrl ints (at 524288): [0..7] counts, [8..15] cursors, [16..24] rowStart, [25..33] groupStart
// bins ints (at 524288+256): N row indices grouped by cluster

__global__ void cpf_zero(int* ctrl) {
    if (threadIdx.x < 64) ctrl[threadIdx.x] = 0;
}

__global__ void cpf_count(const int* __restrict__ labels, int* ctrl, int n) {
    int i = blockIdx.x * blockDim.x + threadIdx.x;
    if (i < n) atomicAdd(&ctrl[labels[i] & 7], 1);
}

__global__ void cpf_prefix(int* ctrl) {
    if (threadIdx.x == 0) {
        int rs = 0, gs = 0;
        ctrl[16] = 0; ctrl[25] = 0;
        for (int c = 0; c < PCLUS; ++c) {
            rs += ctrl[c];
            gs += (ctrl[c] + 127) >> 7;
            ctrl[16 + c + 1] = rs;
            ctrl[25 + c + 1] = gs;
        }
    }
}

__global__ void cpf_fill(const int* __restrict__ labels, int* ctrl, int* bins, int n) {
    int i = blockIdx.x * blockDim.x + threadIdx.x;
    if (i < n) {
        int c = labels[i] & 7;
        int pos = atomicAdd(&ctrl[8 + c], 1);
        bins[ctrl[16 + c] + pos] = i;
    }
}

// Split W[k][n][c] (f32) into bf16 hi/lo planes in B-operand chunk order.
__global__ void cpf_splitW(const float* __restrict__ W, unsigned char* __restrict__ wt) {
    int t = blockIdx.x * blockDim.x + threadIdx.x;   // 8192 threads total
    if (t >= PCLUS * 4 * 2 * CDIM) return;
    int n  = t & 127;
    int h  = (t >> 7) & 1;
    int ks = (t >> 8) & 3;
    int c  = t >> 10;
    union { __bf16 b[16]; uint4 q[2]; } hi, lo;
#pragma unroll
    for (int e = 0; e < 16; ++e) {
        int k = ks * 32 + 16 * h + e;
        float w = W[((size_t)k * CDIM + n) * PCLUS + c];
        __bf16 wh = (__bf16)w;
        hi.b[e] = wh;
        lo.b[e] = (__bf16)(w - (float)wh);
    }
    size_t coff = (size_t)c * 65536 + ((size_t)(ks * 2 + h) * CDIM + n) * 32;
    uint4* dh = (uint4*)(wt + coff);
    uint4* dl = (uint4*)(wt + coff + 32768);
    dh[0] = hi.q[0]; dh[1] = hi.q[1];
    dl[0] = lo.q[0]; dl[1] = lo.q[1];
}

// LDS: Wh 32KB | Wl 32KB | Xh 8*16*272 | Xl 8*16*272  = 135168 bytes
#define SM_WL 32768
#define SM_XH 65536
#define SM_XL 100352
#define SM_TOTAL 135168
#define XROWSTRIDE 272   // 128 bf16 = 256B + 16B pad (bank-conflict mitigation)

__global__ __launch_bounds__(256) void cpf_gemm(const float* __restrict__ x,
                                                const unsigned char* __restrict__ wt,
                                                const int* __restrict__ ctrl,
                                                const int* __restrict__ bins,
                                                float* __restrict__ out)
{
    extern __shared__ unsigned char smem[];
    unsigned char* sWh = smem;
    unsigned char* sXh = smem + SM_XH;
    unsigned char* sXl = smem + SM_XL;

    const int gid = blockIdx.x;
    if (gid >= ctrl[25 + 8]) return;               // past last group
    int c = 0;
    while (c < 7 && gid >= ctrl[25 + c + 1]) c++;  // find cluster for this group
    const int gidLocal = gid - ctrl[25 + c];
    const int cnt      = ctrl[c];
    const int rowStart = ctrl[16 + c];

    const int tid  = threadIdx.x;
    const int lane = tid & 31;
    const int wave = tid >> 5;
    const int r    = lane & 15;
    const int h    = lane >> 4;

    // ---- stage Wh+Wl planes (64KB) into LDS ----
    // CDNA5 async global->LDS (ASYNCcnt): overlaps with the row normalization
    // pass below. Falls back to a sync copy if the builtin is unavailable.
#if __has_builtin(__builtin_amdgcn_global_load_async_to_lds_b128)
    {
        unsigned char* gsrc = (unsigned char*)(wt + (size_t)c * 65536 + tid * 16);
        unsigned char* ldst = sWh + tid * 16;
#pragma unroll
        for (int i = 0; i < 16; ++i) {
            __builtin_amdgcn_global_load_async_to_lds_b128(
                (g_v4i_p)(gsrc + i * 4096),
                (l_v4i_p)(ldst + i * 4096),
                0, 0);
        }
    }
#define CPF_ASYNC_STAGE 1
#else
    {
        const uint4* src = (const uint4*)(wt + (size_t)c * 65536);
        uint4* dst = (uint4*)sWh;
#pragma unroll
        for (int i = 0; i < 16; ++i) dst[tid + i * 256] = src[tid + i * 256];
    }
#endif

    // ---- gather + normalize 16 rows per wave, store bf16 hi/lo to LDS ----
    const int segBase  = gidLocal * 128 + wave * 16;   // position within cluster
    const int segStart = rowStart + segBase;
    const int fallbackRow = bins[rowStart];            // cluster nonempty if group exists
#pragma unroll 1
    for (int rr = 0; rr < 16; ++rr) {
        const bool valid = (segBase + rr) < cnt;
        const int row = valid ? bins[segStart + rr] : fallbackRow;
        float4 v = ((const float4*)(x + (size_t)row * CDIM))[lane];
        float s  = v.x + v.y + v.z + v.w;
        float ss = v.x * v.x + v.y * v.y + v.z * v.z + v.w * v.w;
#pragma unroll
        for (int mk = 1; mk < 32; mk <<= 1) {
            s  += __shfl_xor(s,  mk, 32);
            ss += __shfl_xor(ss, mk, 32);
        }
        const float mean = s * (1.0f / 128.0f);
        const float var  = (ss - 128.0f * mean * mean) * (1.0f / 127.0f);  // ddof=1
        const float rstd = rsqrtf(var + 1.0f);                             // eps = 1
        union { __bf16 b[4]; unsigned long long u; } ph, pl;
        float f0 = (v.x - mean) * rstd;
        float f1 = (v.y - mean) * rstd;
        float f2 = (v.z - mean) * rstd;
        float f3 = (v.w - mean) * rstd;
        __bf16 h0 = (__bf16)f0, h1 = (__bf16)f1, h2 = (__bf16)f2, h3 = (__bf16)f3;
        ph.b[0] = h0; ph.b[1] = h1; ph.b[2] = h2; ph.b[3] = h3;
        pl.b[0] = (__bf16)(f0 - (float)h0);
        pl.b[1] = (__bf16)(f1 - (float)h1);
        pl.b[2] = (__bf16)(f2 - (float)h2);
        pl.b[3] = (__bf16)(f3 - (float)h3);
        const int off = (wave * 16 + rr) * XROWSTRIDE + lane * 8;
        *(unsigned long long*)(sXh + off) = ph.u;
        *(unsigned long long*)(sXl + off) = pl.u;
    }

#ifdef CPF_ASYNC_STAGE
    // All async W-tile transfers must land before any wave reads sWh/sWl.
#if __has_builtin(__builtin_amdgcn_s_wait_asynccnt)
    __builtin_amdgcn_s_wait_asynccnt(0);
#else
    asm volatile("s_wait_asynccnt 0x0" ::: "memory");
#endif
#endif
    __syncthreads();

    // ---- split-bf16 GEMM: acc += xh*Wh + xl*Wh + xh*Wl ----
    v8f acc[8] = {};
#pragma unroll
    for (int ks = 0; ks < 4; ++ks) {
        // A operand: lane-half contiguous K {16h..16h+15} per 32-K step
        union { v16bf v; v8bf p[2]; } ah, al;
        const int abase = (wave * 16 + r) * XROWSTRIDE + ks * 64 + h * 32;
        ah.p[0] = *(const v8bf*)(sXh + abase);
        ah.p[1] = *(const v8bf*)(sXh + abase + 16);
        al.p[0] = *(const v8bf*)(sXl + abase);
        al.p[1] = *(const v8bf*)(sXl + abase + 16);
#pragma unroll
        for (int nt = 0; nt < 8; ++nt) {
            const int n = nt * 16 + r;
            const int boff = ((ks * 2 + h) * CDIM + n) * 32;
            union { v16bf v; v8bf p[2]; } bh, bl;
            bh.p[0] = *(const v8bf*)(sWh + boff);
            bh.p[1] = *(const v8bf*)(sWh + boff + 16);
            bl.p[0] = *(const v8bf*)(sWh + SM_WL + boff);
            bl.p[1] = *(const v8bf*)(sWh + SM_WL + boff + 16);
            acc[nt] = __builtin_amdgcn_wmma_f32_16x16x32_bf16(
                false, ah.v, false, bh.v, (short)0, acc[nt], false, false);
            acc[nt] = __builtin_amdgcn_wmma_f32_16x16x32_bf16(
                false, al.v, false, bh.v, (short)0, acc[nt], false, false);
            acc[nt] = __builtin_amdgcn_wmma_f32_16x16x32_bf16(
                false, ah.v, false, bl.v, (short)0, acc[nt], false, false);
        }
    }

    // ---- tanhshrink epilogue + scatter (D layout: VGPR i -> M=i+8h, N=lane&15) ----
#pragma unroll
    for (int i = 0; i < 8; ++i) {
        const int m = i + 8 * h;
        if ((segBase + m) >= cnt) continue;
        const int row = bins[segStart + m];
        float* orow = out + (size_t)row * CDIM;
#pragma unroll
        for (int nt = 0; nt < 8; ++nt) {
            float vv = acc[nt][i];
            orow[nt * 16 + r] = vv - tanhf(vv);
        }
    }
}

extern "C" void kernel_launch(void* const* d_in, const int* in_sizes, int n_in,
                              void* d_out, int out_size, void* d_ws, size_t ws_size,
                              hipStream_t stream) {
    const float* x      = (const float*)d_in[0];
    const float* W      = (const float*)d_in[1];
    const int*   labels = (const int*)d_in[2];
    float* out = (float*)d_out;
    const int n = in_sizes[0] / CDIM;   // 262144

    unsigned char* wt = (unsigned char*)d_ws;
    int* ctrl = (int*)d_ws + (PCLUS * 65536) / 4;
    int* bins = ctrl + 64;

    cpf_zero  <<<1, 64, 0, stream>>>(ctrl);
    cpf_count <<<(n + 255) / 256, 256, 0, stream>>>(labels, ctrl, n);
    cpf_prefix<<<1, 32, 0, stream>>>(ctrl);
    cpf_fill  <<<(n + 255) / 256, 256, 0, stream>>>(labels, ctrl, bins, n);
    cpf_splitW<<<32, 256, 0, stream>>>(W, wt);

    const int maxGroups = n / 128 + PCLUS;   // upper bound on sum ceil(count_c/128)
    cpf_gemm<<<maxGroups, 256, SM_TOTAL, stream>>>(x, wt, ctrl, bins, out);
}
